// up_idwt_22574348108532
// MI455X (gfx1250) — matrix operand order repaired and checked by reading, not compiled, not run on hardware
//
#include <hip/hip_runtime.h>

// ---------------------------------------------------------------------------
// Types / helpers
// ---------------------------------------------------------------------------
typedef __attribute__((ext_vector_type(16))) __bf16 v16bf;
typedef __attribute__((ext_vector_type(8)))  float  v8f;
typedef __attribute__((ext_vector_type(4)))  int    vi4;
typedef unsigned short bfraw;

union Frag32 { uint4 q[2]; v16bf v; };
union Pack8  { uint4 q; bfraw h[8]; };

__device__ __forceinline__ bfraw f2bf(float f) {
    unsigned u = __float_as_uint(f);
    return (bfraw)((u + 0x7FFFu + ((u >> 16) & 1u)) >> 16);   // RNE
}
__device__ __forceinline__ float bf2f(bfraw h) {
    return __uint_as_float(((unsigned)h) << 16);
}

// CDNA5 async memory->LDS path (ASYNCcnt), guarded so we keep a compiling
// synchronous fallback if this toolchain lacks the builtins.
#if __has_builtin(__builtin_amdgcn_global_load_async_to_lds_b128) && \
    __has_builtin(__builtin_amdgcn_s_wait_asynccnt)
#define USE_ASYNC_LDS 1
#else
#define USE_ASYNC_LDS 0
#endif

typedef __attribute__((address_space(1))) vi4 gvi4;   // global int4
typedef __attribute__((address_space(3))) vi4 lvi4;   // LDS int4

__device__ __forceinline__ gvi4* to_global(const void* p) {
    return (gvi4*)const_cast<void*>(p);
}
__device__ __forceinline__ lvi4* to_lds(void* p) {
    return (lvi4*)p;
}

// ---------------------------------------------------------------------------
// Prep kernels
// ---------------------------------------------------------------------------
__global__ void zero_fill(bfraw* __restrict__ p, int n) {
    int i = blockIdx.x * blockDim.x + threadIdx.x;
    if (i < n) p[i] = 0;
}

__global__ void bn_prep(const float* __restrict__ b, const float* __restrict__ g,
                        const float* __restrict__ be, const float* __restrict__ m,
                        const float* __restrict__ v, float* __restrict__ sc,
                        float* __restrict__ bi, int C) {
    int i = blockIdx.x * blockDim.x + threadIdx.x;
    if (i < C) {
        float s = g[i] * rsqrtf(v[i] + 1e-5f);
        sc[i] = s;
        bi[i] = be[i] + (b[i] - m[i]) * s;
    }
}

// OIHW f32 -> [OC][(ky*3+kx)*128 + ic] bf16  (K tap-major, IC-contiguous)
__global__ void w_to_bf16(const float* __restrict__ src, bfraw* __restrict__ dst, int total) {
    int i = blockIdx.x * 256 + threadIdx.x;
    if (i >= total) return;
    int kx = i % 3; int t = i / 3;
    int ky = t % 3; t /= 3;
    int ic = t % 128; int oc = t / 128;
    dst[((size_t)oc * 9 + ky * 3 + kx) * 128 + ic] = f2bf(src[i]);
}

// NCHW f32 -> NHWC bf16 (channel offset into wider buffer => fused concat)
__global__ void nchw_to_nhwc(const float* __restrict__ src, bfraw* __restrict__ dst,
                             int C, int HW, int Cstride, int chOff, int total) {
    int i = blockIdx.x * 256 + threadIdx.x;
    if (i >= total) return;
    int hw = i % HW; int t = i / HW;
    int c = t % C;  int n = t / C;
    dst[((size_t)n * HW + hw) * Cstride + chOff + c] = f2bf(src[i]);
}

// ---------------------------------------------------------------------------
// Haar IDWT: h NHWC bf16 [16][64][64][512] -> idwt NHWC bf16 [16][128][128][128]
// ---------------------------------------------------------------------------
__global__ __launch_bounds__(256) void haar_idwt(const bfraw* __restrict__ h,
                                                 bfraw* __restrict__ out) {
    int i = blockIdx.x * 256 + threadIdx.x;          // 16*64*64*16 threads
    const int c8 = i & 15; int t = i >> 4;
    const int x = t & 63;  t >>= 6;
    const int y = t & 63;
    const int n = t >> 6;

    const bfraw* base = h + ((size_t)((n * 64 + y) * 64 + x)) * 512 + c8 * 8;
    Pack8 A, Hc, V, D;
    A.q  = *(const uint4*)(base);
    Hc.q = *(const uint4*)(base + 128);
    V.q  = *(const uint4*)(base + 256);
    D.q  = *(const uint4*)(base + 384);

    Pack8 o00, o01, o10, o11;
#pragma unroll
    for (int r = 0; r < 8; ++r) {
        float a  = bf2f(A.h[r]),  hh = bf2f(Hc.h[r]);
        float vv = bf2f(V.h[r]),  dd = bf2f(D.h[r]);
        o00.h[r] = f2bf(0.5f * (a + hh + vv + dd));
        o01.h[r] = f2bf(0.5f * (a + hh - vv - dd));
        o10.h[r] = f2bf(0.5f * (a - hh + vv - dd));
        o11.h[r] = f2bf(0.5f * (a - hh - vv + dd));
    }
    bfraw* ob = out + ((size_t)((n * 128 + 2 * y) * 128 + 2 * x)) * 128 + c8 * 8;
    *(uint4*)(ob)                   = o00.q;
    *(uint4*)(ob + 128)             = o01.q;
    *(uint4*)(ob + 128 * 128)       = o10.q;
    *(uint4*)(ob + 128 * 128 + 128) = o11.q;
}

// ---------------------------------------------------------------------------
// Implicit-GEMM 3x3 conv, IC=128, bf16 WMMA 16x16x32, f32 accum.
//   block: 256 threads = 8 waves (2 M x 4 N), tile 64 OC x 256 px
//   wave : 32 OC x 64 px  -> 8 v_wmma per K-step, acc[2][4]
//   K    : 9 taps * 128 ic = 36 steps of 32, double-buffered LDS,
//          staged with GLOBAL_LOAD_ASYNC_TO_LDS_B128 (ASYNCcnt) when available
//   MODE 0: bf16 NHWC out (Cout stride, chOff) fused BN+ReLU
//   MODE 1: f32 NCHW out (d_out) fused BN+ReLU
// ---------------------------------------------------------------------------
template <int DIM, int MODE>
__global__ __launch_bounds__(256)
void conv3x3_bf16_wmma(const bfraw* __restrict__ Wt,   // [OC][1152] bf16
                       const bfraw* __restrict__ X,    // NHWC bf16, C=128
                       const bfraw* __restrict__ Zb,   // 256B of zeros (OOB src)
                       void* __restrict__ Out,
                       const float* __restrict__ scale,
                       const float* __restrict__ bias,
                       int Cout, int chOff, int OCtot) {
    constexpr int IC = 128;
    constexpr int KS = IC * 9;
    constexpr int HW = DIM * DIM;

    __shared__ __align__(16) bfraw lA[2][64 * 32];     // 2 x 4 KB
    __shared__ __align__(16) bfraw lB[2][256 * 32];    // 2 x 16 KB

    const int tid    = threadIdx.x;
    const int pBase  = blockIdx.x * 256;
    const int ocBase = blockIdx.y * 64;

    // B staging: one thread per pixel, 64 B (4 x b128)
    const int p   = pBase + tid;
    const int n   = p / HW;
    const int rem = p - n * HW;
    const int py  = rem / DIM;
    const int px  = rem - py * DIM;

    // A staging: 4 threads per OC row, 16 B each
    const int aRow   = tid >> 2;
    const int aChunk = tid & 3;
    const bfraw* wRow = Wt + (size_t)(ocBase + aRow) * KS + aChunk * 8;

    const int wid   = tid >> 5;
    const int lane  = tid & 31;
    const int wm    = wid >> 2;       // 0..1 (M)
    const int wn    = wid & 3;        // 0..3 (N), 64 px per wave
    const int lhalf = lane >> 4;
    const int lmod  = lane & 15;

    v8f acc[2][4] = {};

    auto stage = [&](int k0, int buf) {
        const int f  = k0 >> 7;               // tap 0..8 (IC = 128)
        const int ic = k0 & (IC - 1);
        const int ky = f / 3 - 1;
        const int kx = f - (f / 3) * 3 - 1;
        const int yy = py + ky, xx = px + kx;
        const bool inb = (unsigned)yy < (unsigned)DIM && (unsigned)xx < (unsigned)DIM;
        const bfraw* srcB = inb ? X + (((size_t)n * HW + yy * DIM + xx) * IC + ic) : Zb;
        const bfraw* srcA = wRow + k0;
        bfraw* dstA = &lA[buf][aRow * 32 + aChunk * 8];
        bfraw* dstB = &lB[buf][tid * 32];
        __builtin_prefetch(wRow + k0 + 64, 0, 3);
#if USE_ASYNC_LDS
        __builtin_amdgcn_global_load_async_to_lds_b128(to_global(srcA), to_lds(dstA), 0, 0);
#pragma unroll
        for (int q = 0; q < 4; ++q)
            __builtin_amdgcn_global_load_async_to_lds_b128(to_global(srcB + q * 8),
                                                           to_lds(dstB + q * 8), 0, 0);
#else
        uint4 av = *(const uint4*)srcA;
        uint4 bv[4];
#pragma unroll
        for (int q = 0; q < 4; ++q) bv[q] = ((const uint4*)srcB)[q];
        *(uint4*)dstA = av;
#pragma unroll
        for (int q = 0; q < 4; ++q) ((uint4*)dstB)[q] = bv[q];
#endif
    };

    stage(0, 0);

    for (int k0 = 0; k0 < KS; k0 += 32) {
        const int cur = (k0 >> 5) & 1;
#if USE_ASYNC_LDS
        __builtin_amdgcn_s_wait_asynccnt(0);    // staged tile landed in LDS
#endif
        __syncthreads();
        if (k0 + 32 < KS) stage(k0 + 32, cur ^ 1);   // overlap next stage w/ WMMAs

        // A frag: lane = M row; VGPR0-3: K = 8*lhalf..+8, VGPR4-7: K = 16+8*lhalf..+8
        Frag32 afr[2], bfr[4];
#pragma unroll
        for (int sm = 0; sm < 2; ++sm) {
            const int row = wm * 32 + sm * 16 + lmod;
            afr[sm].q[0] = *(const uint4*)&lA[cur][row * 32 + lhalf * 8];
            afr[sm].q[1] = *(const uint4*)&lA[cur][row * 32 + 16 + lhalf * 8];
        }
        // B frag: lane = N col; VGPR0-3: K = 8*lhalf..+8, VGPR4-7: K = 16+8*lhalf..+8
#pragma unroll
        for (int sn = 0; sn < 4; ++sn) {
            const int pix = wn * 64 + sn * 16 + lmod;
            bfr[sn].q[0] = *(const uint4*)&lB[cur][pix * 32 + lhalf * 8];
            bfr[sn].q[1] = *(const uint4*)&lB[cur][pix * 32 + 16 + lhalf * 8];
        }
#pragma unroll
        for (int sm = 0; sm < 2; ++sm)
#pragma unroll
            for (int sn = 0; sn < 4; ++sn)
                acc[sm][sn] = __builtin_amdgcn_wmma_f32_16x16x32_bf16(
                    false, afr[sm].v, false, bfr[sn].v,
                    (short)0, acc[sm][sn], false, false);
    }

    // Epilogue: fused BN (scale/bias) + ReLU; 8 consecutive OCs per lane
#pragma unroll
    for (int sm = 0; sm < 2; ++sm) {
        const int ocL = wm * 32 + sm * 16 + lhalf * 8;
        float sc[8], bi[8];
#pragma unroll
        for (int r = 0; r < 8; ++r) {
            sc[r] = scale[ocBase + ocL + r];
            bi[r] = bias[ocBase + ocL + r];
        }
#pragma unroll
        for (int sn = 0; sn < 4; ++sn) {
            const int pp = pBase + wn * 64 + sn * 16 + lmod;
            float vr[8];
#pragma unroll
            for (int r = 0; r < 8; ++r)
                vr[r] = fmaxf(fmaf(acc[sm][sn][r], sc[r], bi[r]), 0.0f);
            if (MODE == 0) {
                Pack8 pk;
#pragma unroll
                for (int r = 0; r < 8; ++r) pk.h[r] = f2bf(vr[r]);
                bfraw* dst = (bfraw*)Out + (size_t)pp * Cout + chOff + ocBase + ocL;
                *(uint4*)dst = pk.q;
            } else {
                const int nn = pp / HW;
                const int ip = pp - nn * HW;
                float* dst = (float*)Out;
#pragma unroll
                for (int r = 0; r < 8; ++r)
                    dst[((size_t)nn * OCtot + ocBase + ocL + r) * HW + ip] = vr[r];
            }
        }
    }
}

// ---------------------------------------------------------------------------
// Launcher
// ---------------------------------------------------------------------------
extern "C" void kernel_launch(void* const* d_in, const int* in_sizes, int n_in,
                              void* d_out, int out_size, void* d_ws, size_t ws_size,
                              hipStream_t stream) {
    const float* x   = (const float*)d_in[0];
    const float* s   = (const float*)d_in[1];
    const float* w1  = (const float*)d_in[2];
    const float* b1  = (const float*)d_in[3];
    const float* g1  = (const float*)d_in[4];
    const float* be1 = (const float*)d_in[5];
    const float* m1  = (const float*)d_in[6];
    const float* v1  = (const float*)d_in[7];
    const float* wsw = (const float*)d_in[8];
    const float* bs  = (const float*)d_in[9];
    const float* gs  = (const float*)d_in[10];
    const float* bes = (const float*)d_in[11];
    const float* ms  = (const float*)d_in[12];
    const float* vs  = (const float*)d_in[13];
    const float* w2  = (const float*)d_in[14];
    const float* b2  = (const float*)d_in[15];
    const float* g2  = (const float*)d_in[16];
    const float* be2 = (const float*)d_in[17];
    const float* m2  = (const float*)d_in[18];
    const float* v2  = (const float*)d_in[19];

    char* wsbase = (char*)d_ws;
    size_t off = 0;
    auto alloc = [&](size_t bytes) -> char* {
        off = (off + 255) & ~(size_t)255;
        char* ptr = wsbase + off;
        off += bytes;
        return ptr;
    };

    bfraw* xb   = (bfraw*)alloc((size_t)16 * 4096 * 128 * 2);   // x  NHWC bf16
    bfraw* w1b  = (bfraw*)alloc((size_t)512 * 1152 * 2);
    bfraw* wsb  = (bfraw*)alloc((size_t)64 * 1152 * 2);
    bfraw* w2b  = (bfraw*)alloc((size_t)64 * 1152 * 2);
    bfraw* hb   = (bfraw*)alloc((size_t)16 * 4096 * 512 * 2);   // conv1 out NHWC
    bfraw* idwb = (bfraw*)alloc((size_t)16 * 16384 * 128 * 2);  // idwt  NHWC
    bfraw* catb = (bfraw*)alloc((size_t)16 * 16384 * 128 * 2);  // [s | s_out] NHWC
    bfraw* zb   = (bfraw*)alloc(256);                           // OOB zero source
    float* sc1 = (float*)alloc(512 * 4);
    float* bi1 = (float*)alloc(512 * 4);
    float* scS = (float*)alloc(64 * 4);
    float* biS = (float*)alloc(64 * 4);
    float* sc2 = (float*)alloc(64 * 4);
    float* bi2 = (float*)alloc(64 * 4);

    zero_fill<<<1, 128, 0, stream>>>(zb, 128);

    bn_prep<<<2, 256, 0, stream>>>(b1, g1, be1, m1, v1, sc1, bi1, 512);
    bn_prep<<<1, 64, 0, stream>>>(bs, gs, bes, ms, vs, scS, biS, 64);
    bn_prep<<<1, 64, 0, stream>>>(b2, g2, be2, m2, v2, sc2, bi2, 64);

    w_to_bf16<<<(512 * 1152 + 255) / 256, 256, 0, stream>>>(w1, w1b, 512 * 1152);
    w_to_bf16<<<(64 * 1152 + 255) / 256, 256, 0, stream>>>(wsw, wsb, 64 * 1152);
    w_to_bf16<<<(64 * 1152 + 255) / 256, 256, 0, stream>>>(w2, w2b, 64 * 1152);

    nchw_to_nhwc<<<(16 * 128 * 4096 + 255) / 256, 256, 0, stream>>>(
        x, xb, 128, 4096, 128, 0, 16 * 128 * 4096);
    nchw_to_nhwc<<<(16 * 64 * 16384 + 255) / 256, 256, 0, stream>>>(
        s, catb, 64, 16384, 128, 0, 16 * 64 * 16384);

    // conv1 (128->512, 64x64) + BN + ReLU -> h (NHWC bf16)
    conv3x3_bf16_wmma<64, 0><<<dim3(256, 8), 256, 0, stream>>>(
        w1b, xb, zb, hb, sc1, bi1, 512, 0, 512);

    // Haar IDWT: h -> idwt (128ch, 128x128, NHWC bf16)
    haar_idwt<<<4096, 256, 0, stream>>>(hb, idwb);

    // conv_s (128->64, 128x128) + BN + ReLU -> channels [64,128) of concat buf
    conv3x3_bf16_wmma<128, 0><<<dim3(1024, 1), 256, 0, stream>>>(
        wsb, idwb, zb, catb, scS, biS, 128, 64, 64);

    // conv2 (128->64, 128x128) + BN + ReLU -> d_out (f32 NCHW)
    conv3x3_bf16_wmma<128, 1><<<dim3(1024, 1), 256, 0, stream>>>(
        w2b, catb, zb, d_out, sc2, bi2, 64, 0, 64);
}
